// MultiHeadAttention_20418274525947
// MI455X (gfx1250) — matrix operand order repaired and checked by reading, not compiled, not run on hardware
//
#include <hip/hip_runtime.h>
#include <hip/hip_bf16.h>

// ---------------- problem constants ----------------
static constexpr int Bb  = 2;
static constexpr int Ss  = 2048;
static constexpr int Dd  = 1024;
static constexpr int Hh  = 16;
static constexpr int HDd = 64;
static constexpr int Mm  = Bb * Ss;        // 4096 rows

typedef unsigned short u16;
typedef unsigned int   u32;

typedef __attribute__((ext_vector_type(16))) __bf16 bf16x16;
typedef __attribute__((ext_vector_type(8)))  float  f32x8;
typedef __attribute__((ext_vector_type(4)))  u32    u32x4;

union FragU { u32x4 q[2]; bf16x16 v; };

// RNE fp32 -> bf16 (finite inputs)
__device__ __forceinline__ u16 f2bf(float f) {
  u32 u = __float_as_uint(f);
  u32 r = u + 0x7FFFu + ((u >> 16) & 1u);
  return (u16)(r >> 16);
}

// ---- WMMA fragment loaders (CDNA5 16-bit layouts, wave32) ----
// A 16x32 (MxK): lane L -> row M=L&15, K in {kb..kb+7, kb+16..kb+23}, kb=(L<16)?0:8
__device__ __forceinline__ bf16x16 load_a_frag(const u16* __restrict__ base, int ld, int lane) {
  const int r  = lane & 15;
  const int kb = (lane < 16) ? 0 : 8;
  const u16* p = base + (size_t)r * ld + kb;
  FragU u;
  u.q[0] = *(const u32x4*)(p);
  u.q[1] = *(const u32x4*)(p + 16);
  return u.v;
}

// B 32x16 (KxN) from a [N,K] row-major array: lane L -> col N=L&15,
// K in {kb..kb+7, kb+16..kb+23}, kb = (L<16)?0:8
__device__ __forceinline__ bf16x16 load_b_frag(const u16* __restrict__ base, int ld, int lane) {
  const int n  = lane & 15;
  const int kb = (lane < 16) ? 0 : 8;
  const u16* p = base + (size_t)n * ld + kb;
  FragU u;
  u.q[0] = *(const u32x4*)(p);
  u.q[1] = *(const u32x4*)(p + 16);
  return u.v;
}

// A-frag from LDS tile (16 rows x 32 cols of u16)
__device__ __forceinline__ bf16x16 load_a_frag_lds(const u16* base, int lane) {
  const int r  = lane & 15;
  const int kb = (lane < 16) ? 0 : 8;
  const u16* p = base + r * 32 + kb;
  FragU u;
  u.q[0] = *(const u32x4*)(p);
  u.q[1] = *(const u32x4*)(p + 16);
  return u.v;
}

__device__ __forceinline__ f32x8 wmma_bf16(bf16x16 a, bf16x16 b, f32x8 c) {
  return __builtin_amdgcn_wmma_f32_16x16x32_bf16(false, a, false, b, (short)0, c, false, false);
}

// ---- row-max reduction over 16-lane halves (VALU permlane path if available) ----
#if __has_builtin(__builtin_amdgcn_permlane16)
__device__ __forceinline__ float pl16(float v, u32 s0, u32 s1) {
  return __int_as_float((int)__builtin_amdgcn_permlane16(
      (unsigned)__float_as_int(v), (unsigned)__float_as_int(v), s0, s1, false, false));
}
__device__ __forceinline__ float half_reduce_max(float v) {
  v = fmaxf(v, pl16(v, 0x67452301u, 0xEFCDAB89u)); // xor 1
  v = fmaxf(v, pl16(v, 0x45670123u, 0xCDEF89ABu)); // xor 2
  v = fmaxf(v, pl16(v, 0x01234567u, 0x89ABCDEFu)); // xor 4
  v = fmaxf(v, pl16(v, 0xFEDCBA98u, 0x76543210u)); // xor 8
  return v;
}
#else
__device__ __forceinline__ float half_reduce_max(float v) {
  v = fmaxf(v, __shfl_xor(v, 1, 32));
  v = fmaxf(v, __shfl_xor(v, 2, 32));
  v = fmaxf(v, __shfl_xor(v, 4, 32));
  v = fmaxf(v, __shfl_xor(v, 8, 32));
  return v;
}
#endif

// ---------------- fp32 -> bf16 convert (x4 vectorized) ----------------
__global__ void cvt_f32_bf16(const float4* __restrict__ in, uint2* __restrict__ out, int n4) {
  int i = blockIdx.x * blockDim.x + threadIdx.x;
  if (i < n4) {
    float4 v = in[i];
    uint2 r;
    r.x = (u32)f2bf(v.x) | ((u32)f2bf(v.y) << 16);
    r.y = (u32)f2bf(v.z) | ((u32)f2bf(v.w) << 16);
    out[i] = r;
  }
}

// ---------------- GEMM: C[M,N] = oscale * (A[M,K] * W[N,K]^T) ----------------
// MODE 0: store bf16 as Q/K layout [B,H,S,HD]   (oscale applied; Q uses 1/sqrt(hd))
// MODE 1: store bf16 as V^T layout [B,H,HD,S]
// MODE 2: store fp32 row-major [M,N] with bias
template <int MODE>
__global__ void gemm_bf16_k(const u16* __restrict__ A, const u16* __restrict__ W,
                            void* __restrict__ Out, const float* __restrict__ bias,
                            int M, int N, int K, float oscale) {
  const int lane = threadIdx.x & 31;
  const int wave = threadIdx.x >> 5;
  const int m0 = blockIdx.x * 128 + wave * 16;   // 8 waves x 16 rows
  const int n0 = blockIdx.y * 64;                // 4 x 16 cols per wave

  f32x8 acc[4];
  #pragma unroll
  for (int t = 0; t < 4; ++t) acc[t] = (f32x8)(0.0f);

  const u16* Abase = A + (size_t)m0 * K;
  for (int k = 0; k < K; k += 32) {
    bf16x16 af = load_a_frag(Abase + k, K, lane);
    #pragma unroll
    for (int t = 0; t < 4; ++t) {
      bf16x16 bf = load_b_frag(W + (size_t)(n0 + t * 16) * K + k, K, lane);
      acc[t] = wmma_bf16(af, bf, acc[t]);
    }
  }

  const int half = lane >> 4;
  const int colL = lane & 15;
  #pragma unroll
  for (int t = 0; t < 4; ++t) {
    #pragma unroll
    for (int i = 0; i < 8; ++i) {
      const int row = m0 + i + 8 * half;   // global m (token)
      const int col = n0 + t * 16 + colL;  // global n (feature)
      const float v = acc[t][i];
      if (MODE == 2) {
        ((float*)Out)[(size_t)row * N + col] = v + bias[col];
      } else {
        const int b  = row / Ss, s = row % Ss;
        const int h  = col >> 6, hd = col & 63;
        const u16 bv = f2bf(v * oscale);
        if (MODE == 0)
          ((u16*)Out)[(((size_t)b * Hh + h) * Ss + s) * HDd + hd] = bv;
        else
          ((u16*)Out)[(((size_t)b * Hh + h) * HDd + hd) * Ss + s] = bv;
      }
    }
  }
}

// ---------------- flash attention tile step ----------------
// Q pre-scaled by 1/sqrt(HD). MASKED only for the diagonal tile.
template <bool MASKED>
__device__ __forceinline__ void attn_tile(
    int k0, int q0, int lane, int rlb, int colL,
    const u16* __restrict__ Kb, const u16* __restrict__ Vb, u16* lds,
    bf16x16 qf0, bf16x16 qf1, bf16x16 ones,
    f32x8& acc0, f32x8& acc1, f32x8& acc2, f32x8& acc3,
    float (&rowM)[8], float (&rowL)[8]) {
  // ---- scores S = Q K^T  (16 x 32), 4 WMMAs ----
  f32x8 s0 = (f32x8)(0.0f), s1 = (f32x8)(0.0f);
  {
    bf16x16 b00 = load_b_frag(Kb + (size_t)k0 * HDd, HDd, lane);
    bf16x16 b01 = load_b_frag(Kb + (size_t)k0 * HDd + 32, HDd, lane);
    s0 = wmma_bf16(qf0, b00, s0);
    s0 = wmma_bf16(qf1, b01, s0);
    bf16x16 b10 = load_b_frag(Kb + (size_t)(k0 + 16) * HDd, HDd, lane);
    bf16x16 b11 = load_b_frag(Kb + (size_t)(k0 + 16) * HDd + 32, HDd, lane);
    s1 = wmma_bf16(qf0, b10, s1);
    s1 = wmma_bf16(qf1, b11, s1);
  }

  // ---- online softmax: max + exp, stage P to LDS ----
  float corr[8];
  const int kc0 = k0 + colL;
  #pragma unroll
  for (int i = 0; i < 8; ++i) {
    float v0 = s0[i];
    float v1 = s1[i];
    if (MASKED) {
      const int qi = q0 + rlb + i;
      if (kc0 > qi)      v0 = -3.0e38f;
      if (kc0 + 16 > qi) v1 = -3.0e38f;
    }
    const float mx = half_reduce_max(fmaxf(v0, v1));
    const float nM = fmaxf(rowM[i], mx);
    corr[i] = __expf(rowM[i] - nM);
    rowM[i] = nM;
    const int row = rlb + i;
    lds[row * 32 + colL]      = f2bf(__expf(v0 - nM));
    lds[row * 32 + colL + 16] = f2bf(__expf(v1 - nM));
  }
  __builtin_amdgcn_wave_barrier();
  const bf16x16 pf = load_a_frag_lds(lds, lane);
  __builtin_amdgcn_wave_barrier();

  // ---- row-sum of P via WMMA against ones (replaces 32 lane shuffles) ----
  f32x8 rs = wmma_bf16(pf, ones, (f32x8)(0.0f));
  #pragma unroll
  for (int i = 0; i < 8; ++i) {
    rowL[i] = rowL[i] * corr[i] + rs[i];
    acc0[i] *= corr[i]; acc1[i] *= corr[i]; acc2[i] *= corr[i]; acc3[i] *= corr[i];
  }

  // ---- O += P * V   (B-frags from Vt[hd, key], contiguous keys) ----
  acc0 = wmma_bf16(pf, load_b_frag(Vb + (size_t)(0 * 16) * Ss + k0, Ss, lane), acc0);
  acc1 = wmma_bf16(pf, load_b_frag(Vb + (size_t)(1 * 16) * Ss + k0, Ss, lane), acc1);
  acc2 = wmma_bf16(pf, load_b_frag(Vb + (size_t)(2 * 16) * Ss + k0, Ss, lane), acc2);
  acc3 = wmma_bf16(pf, load_b_frag(Vb + (size_t)(3 * 16) * Ss + k0, Ss, lane), acc3);
}

// ---------------- flash attention: one wave per 16-query tile ----------------
// Q,K: [B,H,S,64] bf16 (Q pre-scaled).  Vt: [B,H,64,S] bf16.  Ctx: [B,S,D] bf16.
__global__ void flash_attn_k(const u16* __restrict__ Q, const u16* __restrict__ Kmat,
                             const u16* __restrict__ Vt, u16* __restrict__ Ctx) {
  __shared__ u16 smem[8][16 * 32];

  const int lane = threadIdx.x & 31;
  const int wave = threadIdx.x >> 5;
  const int id   = blockIdx.x * 8 + wave;        // global wave-tile id
  const int QT   = Ss / 16;                      // 128 q-tiles per (b,h)
  const int qt   = id % QT;
  const int bh   = id / QT;                      // b*H + h
  const int q0   = qt * 16;

  const u16* Qb = Q    + (size_t)bh * Ss * HDd + (size_t)q0 * HDd;
  const u16* Kb = Kmat + (size_t)bh * Ss * HDd;
  const u16* Vb = Vt   + (size_t)bh * HDd * Ss;
  u16* lds = &smem[wave][0];

  const bf16x16 qf0 = load_a_frag(Qb, HDd, lane);       // hd 0..31
  const bf16x16 qf1 = load_a_frag(Qb + 32, HDd, lane);  // hd 32..63

  bf16x16 ones;                                   // all-1.0 bf16 B fragment
  {
    FragU u;
    u.q[0] = (u32x4)(0x3F803F80u);
    u.q[1] = (u32x4)(0x3F803F80u);
    ones = u.v;
  }

  f32x8 acc0 = (f32x8)(0.0f), acc1 = (f32x8)(0.0f), acc2 = (f32x8)(0.0f), acc3 = (f32x8)(0.0f);
  float rowM[8], rowL[8];
  #pragma unroll
  for (int i = 0; i < 8; ++i) { rowM[i] = -3.0e38f; rowL[i] = 0.0f; }

  const int rlb  = (lane >> 4) * 8;   // row_local base (0 or 8)
  const int colL = lane & 15;
  const int jmax = (q0 + 15) / 32;    // inclusive; 32-key tiles; tile jmax is diagonal

  // bulk tiles: fully below the causal diagonal, no masking
  for (int j = 0; j < jmax; ++j) {
    const int k0 = j * 32;
    // per-lane prefetch of next K/V tiles (global_prefetch_b8)
    __builtin_prefetch((const void*)(Kb + (size_t)(k0 + 32 + lane) * HDd), 0, 1);
    __builtin_prefetch((const void*)(Vb + (size_t)lane * Ss + k0 + 32), 0, 1);
    __builtin_prefetch((const void*)(Vb + (size_t)(32 + lane) * Ss + k0 + 32), 0, 1);
    attn_tile<false>(k0, q0, lane, rlb, colL, Kb, Vb, lds, qf0, qf1, ones,
                     acc0, acc1, acc2, acc3, rowM, rowL);
  }
  // diagonal tile with causal mask
  attn_tile<true>(jmax * 32, q0, lane, rlb, colL, Kb, Vb, lds, qf0, qf1, ones,
                  acc0, acc1, acc2, acc3, rowM, rowL);

  // ---- normalize and store ctx (merged heads, [B,S,D]) ----
  const int b = bh >> 4;
  const int h = bh & 15;
  #pragma unroll
  for (int i = 0; i < 8; ++i) {
    const float inv = 1.0f / rowL[i];
    const int s = q0 + rlb + i;
    const size_t ro = ((size_t)b * Ss + s) * Dd + (size_t)h * HDd;
    Ctx[ro + 0 * 16 + colL] = f2bf(acc0[i] * inv);
    Ctx[ro + 1 * 16 + colL] = f2bf(acc1[i] * inv);
    Ctx[ro + 2 * 16 + colL] = f2bf(acc2[i] * inv);
    Ctx[ro + 3 * 16 + colL] = f2bf(acc3[i] * inv);
  }
}

// ---------------- launch ----------------
extern "C" void kernel_launch(void* const* d_in, const int* in_sizes, int n_in,
                              void* d_out, int out_size, void* d_ws, size_t ws_size,
                              hipStream_t stream) {
  const float* x  = (const float*)d_in[0];
  const float* wq = (const float*)d_in[1];
  const float* wk = (const float*)d_in[2];
  const float* wv = (const float*)d_in[3];
  const float* wo = (const float*)d_in[4];
  const float* bo = (const float*)d_in[5];

  // workspace layout (bf16 elements); total ~50 MB
  u16* p   = (u16*)d_ws;
  u16* xb  = p; p += (size_t)Mm * Dd;        // 4096x1024
  u16* wqb = p; p += (size_t)Dd * Dd;
  u16* wkb = p; p += (size_t)Dd * Dd;
  u16* wvb = p; p += (size_t)Dd * Dd;
  u16* wob = p; p += (size_t)Dd * Dd;
  u16* Qb  = p; p += (size_t)Mm * Dd;        // [B,H,S,64] (pre-scaled by 1/8)
  u16* Kb  = p; p += (size_t)Mm * Dd;        // [B,H,S,64]
  u16* Vtb = p; p += (size_t)Mm * Dd;        // [B,H,64,S]
  u16* Ctx = p; p += (size_t)Mm * Dd;        // [B,S,D]

  // 1) convert inputs to bf16
  {
    int n4 = (Mm * Dd) / 4;
    cvt_f32_bf16<<<(n4 + 255) / 256, 256, 0, stream>>>((const float4*)x, (uint2*)xb, n4);
    int w4 = (Dd * Dd) / 4;
    cvt_f32_bf16<<<(w4 + 255) / 256, 256, 0, stream>>>((const float4*)wq, (uint2*)wqb, w4);
    cvt_f32_bf16<<<(w4 + 255) / 256, 256, 0, stream>>>((const float4*)wk, (uint2*)wkb, w4);
    cvt_f32_bf16<<<(w4 + 255) / 256, 256, 0, stream>>>((const float4*)wv, (uint2*)wvb, w4);
    cvt_f32_bf16<<<(w4 + 255) / 256, 256, 0, stream>>>((const float4*)wo, (uint2*)wob, w4);
  }

  // 2) projections (WMMA GEMMs); Q pre-scaled by 1/sqrt(64)=0.125 (exact in bf16)
  dim3 ggrid(Mm / 128, Dd / 64);
  gemm_bf16_k<0><<<ggrid, 256, 0, stream>>>(xb, wqb, Qb,  nullptr, Mm, Dd, Dd, 0.125f);
  gemm_bf16_k<0><<<ggrid, 256, 0, stream>>>(xb, wkb, Kb,  nullptr, Mm, Dd, Dd, 1.0f);
  gemm_bf16_k<1><<<ggrid, 256, 0, stream>>>(xb, wvb, Vtb, nullptr, Mm, Dd, Dd, 1.0f);

  // 3) flash attention: B*H*(S/16) = 4096 wave-tiles, 8 waves/block
  flash_attn_k<<<(Bb * Hh * (Ss / 16)) / 8, 256, 0, stream>>>(Qb, Kb, Vtb, Ctx);

  // 4) output projection + bias (fp32 out)
  gemm_bf16_k<2><<<ggrid, 256, 0, stream>>>(Ctx, wob, d_out, bo, Mm, Dd, Dd, 1.0f);
}
